// EquivariantRotationConv_12515534700845
// MI455X (gfx1250) — matrix operand order repaired and checked by reading, not compiled, not run on hardware
//
#include <hip/hip_runtime.h>
#include <math.h>

// ---------------------------------------------------------------------------
// EquivariantRotationConv for MI455X (gfx1250), bf16 WMMA implicit GEMM.
//
//  d_in[0]: x          (16,128,128,128) f32
//  d_in[1]: weight     (1024,128,3,3)   f32   [(o*8+r), i, ky, kx]
//  d_in[2]: rot_alpha  (8,)             f32
//  d_out  : (16,128,128,128) f32  = max over 8 rotations of 3x3 conv
//
//  Workspace:
//    wsA: rotated filters, bf16, layout [r][tap9][cich4][mtile8][m16][k32]
//         size = 8*9*4*8*512*2 = 2,359,296 B  (L2-resident, reused by all)
//    wsX: x in padded NHWC bf16 [b][py130][px130][ci128]
//         size = 16*130*130*128*2 = 69,222,400 B
//
//  Conv kernel: one block per (batch, output row). 512 threads = 16 waves.
//  Wave w: out-channels [ (w&7)*16, +16 ), pixels [ (w>>3)*64, +64 ) as
//  4 n-tiles. Rotations processed in 2 passes of 4 to keep acc at 128 VGPRs.
//  Per (tap,cich) step: 16 wmma vs 8 global + 8 ds loads.
// ---------------------------------------------------------------------------

typedef __attribute__((ext_vector_type(16))) __bf16 v16bf;
typedef __attribute__((ext_vector_type(8)))  float  v8f;

#define IN_CH   128
#define OUT_CH  128
#define NUM_ROT 8
#define IMG_H   128
#define IMG_W   128
#define BATCH   16
#define PAD_H   130
#define PAD_W   130

#define WSA_ELEMS (NUM_ROT * 9 * 4 * 8 * 512)              // 1,179,648 bf16
#define WSA_BYTES (WSA_ELEMS * 2)                          // 2,359,296
#define WSX_ELEMS ((size_t)BATCH * PAD_H * PAD_W * IN_CH)  // 34,611,200 bf16
#define WSX_BYTES (WSX_ELEMS * 2)

#define A_R_STRIDE   ((size_t)(9 * 4 * 8) * 512)           // bf16 elems per r
#define A_TAP_STRIDE ((size_t)(4 * 8) * 512)               // bf16 elems per tap
#define A_CI_STRIDE  ((size_t)8 * 512)                     // bf16 elems per cich

// ---------------------------------------------------------------------------
// Kernel 1: rotate filter banks (bilinear, align_corners=True, zero pad),
// convert to bf16, scatter into WMMA A-fragment layout.
// ---------------------------------------------------------------------------
__global__ __launch_bounds__(256) void rotfilt_kernel(
    const float* __restrict__ weight, const float* __restrict__ rot_alpha,
    __bf16* __restrict__ wsA)
{
    int tid = blockIdx.x * 256 + threadIdx.x;
    const int total = NUM_ROT * 9 * OUT_CH * IN_CH;
    if (tid >= total) return;

    int i   = tid & (IN_CH - 1);
    int o   = (tid >> 7) & (OUT_CH - 1);
    int tap = (tid >> 14) % 9;
    int r   = tid / (9 * OUT_CH * IN_CH);

    float ang = rot_alpha[r] * 0.78539816339744830962f * (float)r;
    float c, s;
    __sincosf(ang, &s, &c);

    int ky = tap / 3, kx = tap % 3;
    float gy = (float)(ky - 1);          // lin = {-1, 0, 1}
    float gx = (float)(kx - 1);
    float xs = c * gx - s * gy;
    float ys = s * gx + c * gy;
    float ix = xs + 1.0f;                // (xs+1)*0.5*(k-1), k=3
    float iy = ys + 1.0f;
    float x0f = floorf(ix), y0f = floorf(iy);
    int   x0 = (int)x0f,    y0 = (int)y0f;
    float wx = ix - x0f,    wy = iy - y0f;

    const float* wb = weight + (size_t)((o * NUM_ROT + r) * IN_CH + i) * 9;

    float val = 0.0f;
    {
        int yis[4] = { y0, y0,     y0 + 1, y0 + 1 };
        int xis[4] = { x0, x0 + 1, x0,     x0 + 1 };
        float ws[4] = { (1.f - wy) * (1.f - wx), (1.f - wy) * wx,
                        wy * (1.f - wx),          wy * wx };
        #pragma unroll
        for (int t = 0; t < 4; ++t) {
            int yi = yis[t], xi = xis[t];
            bool valid = (yi >= 0) & (yi < 3) & (xi >= 0) & (xi < 3);
            int yc = min(max(yi, 0), 2), xc = min(max(xi, 0), 2);
            val += wb[yc * 3 + xc] * (valid ? ws[t] : 0.0f);
        }
    }

    int blk = ((r * 9 + tap) * 4 + (i >> 5)) * 8 + (o >> 4);
    wsA[(size_t)blk * 512 + (o & 15) * 32 + (i & 31)] = (__bf16)val;
}

// ---------------------------------------------------------------------------
// Kernel 2: zero-fill wsX (provides the zero halo for conv padding).
// ---------------------------------------------------------------------------
__global__ __launch_bounds__(256) void xzero_kernel(uint4* __restrict__ wsX4, int n4)
{
    int tid = blockIdx.x * 256 + threadIdx.x;
    if (tid < n4) wsX4[tid] = make_uint4(0u, 0u, 0u, 0u);
}

// ---------------------------------------------------------------------------
// Kernel 3: NCHW f32 -> padded NHWC bf16 via LDS tile transpose.
// ---------------------------------------------------------------------------
__global__ __launch_bounds__(256) void xprep_kernel(
    const float* __restrict__ x, __bf16* __restrict__ wsX)
{
    __shared__ float tile[32][33];
    int bid  = blockIdx.x;
    int xseg = bid & 3;
    int cich = (bid >> 2) & 3;
    int y    = (bid >> 4) & (IMG_H - 1);
    int b    = bid >> 11;
    int t = threadIdx.x;

    #pragma unroll
    for (int it = 0; it < 4; ++it) {
        int idx = t + it * 256;
        int ci = idx >> 5, px = idx & 31;
        tile[ci][px] = x[(((size_t)b * IN_CH + cich * 32 + ci) * IMG_H + y) * IMG_W
                         + xseg * 32 + px];
    }
    __syncthreads();
    #pragma unroll
    for (int it = 0; it < 4; ++it) {
        int idx = t + it * 256;
        int px = idx >> 5, ci = idx & 31;
        wsX[(((size_t)b * PAD_H + (y + 1)) * PAD_W + (xseg * 32 + px + 1)) * IN_CH
            + cich * 32 + ci] = (__bf16)tile[ci][px];
    }
}

// ---------------------------------------------------------------------------
// Kernel 4: conv + max over rotations.
// ---------------------------------------------------------------------------
__global__ __launch_bounds__(512) void conv_kernel(
    const __bf16* __restrict__ wsA, const __bf16* __restrict__ wsX,
    float* __restrict__ out)
{
    // full padded row patch: 3 rows x 130 px x 128 ci bf16 = 99,840 B
    __shared__ __align__(16) __bf16 xtile[3 * PAD_W * IN_CH];

    int bid = blockIdx.x;
    int y   = bid & (IMG_H - 1);
    int b   = bid >> 7;
    int t   = threadIdx.x;

    // --- cooperative load: 3 padded rows, each 130*128 bf16 = 33,280 B -----
    {
        const int CPR = (PAD_W * IN_CH * 2) / 16;          // 2080 chunks/row
        #pragma unroll 1
        for (int row = 0; row < 3; ++row) {
            const uint4* src = (const uint4*)(wsX
                + ((size_t)b * PAD_H + (y + row)) * PAD_W * IN_CH);
            uint4* dst = (uint4*)(xtile + row * PAD_W * IN_CH);
            for (int off = t; off < CPR; off += 512)
                dst[off] = src[off];
        }
    }
    __syncthreads();

    int wave  = t >> 5, lane = t & 31;
    int m16   = lane & 15, half = lane >> 4;
    int mtile = wave & 7;
    int nbase = (wave >> 3) * 64;        // 0 or 64: this wave's 64-px span

    union Frag { uint4 u[2]; v16bf v; };

    v8f omax[4];
    #pragma unroll
    for (int nt = 0; nt < 4; ++nt)
        #pragma unroll
        for (int j = 0; j < 8; ++j) omax[nt][j] = -__builtin_inff();

    // Per-lane A base: A 16x32 in contiguous k order (same as B):
    // lane (m = lane&15, h = lane>>4) holds k = h*16..h*16+15
    const __bf16* aLane0 = wsA + (size_t)mtile * 512 + m16 * 32 + half * 16;

    #pragma unroll 1
    for (int rpass = 0; rpass < 2; ++rpass) {
        v8f acc[4][4];
        #pragma unroll
        for (int r = 0; r < 4; ++r)
            #pragma unroll
            for (int nt = 0; nt < 4; ++nt)
                #pragma unroll
                for (int j = 0; j < 8; ++j) acc[r][nt][j] = 0.0f;

        const __bf16* aPass = aLane0 + (size_t)rpass * 4 * A_R_STRIDE;

        #pragma unroll 1
        for (int tap = 0; tap < 9; ++tap) {
            int dy = tap / 3, dx = tap - dy * 3;
            const __bf16* aTap = aPass + (size_t)tap * A_TAP_STRIDE;
            if (tap < 8) __builtin_prefetch(aTap + A_TAP_STRIDE, 0, 0);

            // B 32x16 layout: lanes 0-15 hold K=0..15, lanes 16-31 K=16..31,
            // column n = nbase + nt*16 + (lane&15); pixel px = n + dx.
            const __bf16* bBase = xtile
                + ((dy * PAD_W + dx + nbase + m16) * IN_CH + half * 16);

            #pragma unroll 1
            for (int cich = 0; cich < 4; ++cich) {
                Frag bf[4];
                #pragma unroll
                for (int nt = 0; nt < 4; ++nt) {
                    const __bf16* bp = bBase + nt * 16 * IN_CH + cich * 32;
                    bf[nt].u[0] = ((const uint4*)bp)[0];
                    bf[nt].u[1] = ((const uint4*)bp)[1];
                }
                const __bf16* ap = aTap + (size_t)cich * A_CI_STRIDE;
                #pragma unroll
                for (int r = 0; r < 4; ++r) {
                    Frag a;
                    const __bf16* arp = ap + (size_t)r * A_R_STRIDE;
                    a.u[0] = *(const uint4*)(arp);
                    a.u[1] = *(const uint4*)(arp + 8);
                    #pragma unroll
                    for (int nt = 0; nt < 4; ++nt)
                        acc[r][nt] = __builtin_amdgcn_wmma_f32_16x16x32_bf16(
                            false, a.v, false, bf[nt].v, (short)0,
                            acc[r][nt], false, false);
                }
            }
        }

        // fold this pass's rotations into the running max
        #pragma unroll
        for (int r = 0; r < 4; ++r)
            #pragma unroll
            for (int nt = 0; nt < 4; ++nt)
                #pragma unroll
                for (int j = 0; j < 8; ++j)
                    omax[nt][j] = fmaxf(omax[nt][j], acc[r][nt][j]);
    }

    // C/D layout: VGPR j, lanes 0-15 -> M=j, N=lane; lanes 16-31 -> M=8+j.
    #pragma unroll
    for (int nt = 0; nt < 4; ++nt) {
        #pragma unroll
        for (int j = 0; j < 8; ++j) {
            int o = mtile * 16 + half * 8 + j;
            size_t base = (((size_t)b * OUT_CH + o) * IMG_H + y) * IMG_W
                          + nbase + nt * 16 + m16;
            out[base] = omax[nt][j];
        }
    }
}

// ---------------------------------------------------------------------------
extern "C" void kernel_launch(void* const* d_in, const int* in_sizes, int n_in,
                              void* d_out, int out_size, void* d_ws, size_t ws_size,
                              hipStream_t stream)
{
    const float* x         = (const float*)d_in[0];
    const float* weight    = (const float*)d_in[1];
    const float* rot_alpha = (const float*)d_in[2];
    float*       out       = (float*)d_out;

    __bf16* wsA = (__bf16*)d_ws;
    __bf16* wsX = (__bf16*)((char*)d_ws + WSA_BYTES);

    // 1) rotated filters -> WMMA A-fragment layout
    {
        int total = NUM_ROT * 9 * OUT_CH * IN_CH;          // 1,179,648
        rotfilt_kernel<<<(total + 255) / 256, 256, 0, stream>>>(weight, rot_alpha, wsA);
    }
    // 2) zero-fill padded NHWC x (halo = conv zero padding)
    {
        int n4 = (int)(WSX_BYTES / 16);                    // 4,326,400
        xzero_kernel<<<(n4 + 255) / 256, 256, 0, stream>>>((uint4*)wsX, n4);
    }
    // 3) NCHW f32 -> padded NHWC bf16
    {
        int blocks = BATCH * IMG_H * 4 * 4;                // 32768
        xprep_kernel<<<blocks, 256, 0, stream>>>(x, wsX);
    }
    // 4) WMMA conv + rotation max, one block per (batch, row)
    {
        int blocks = BATCH * IMG_H;                        // 2048
        conv_kernel<<<blocks, 512, 0, stream>>>(wsA, wsX, out);
    }
}
